// CustomMaskRCNN_68143951118476
// MI455X (gfx1250) — matrix operand order repaired and checked by reading, not compiled, not run on hardware
//
#include <hip/hip_runtime.h>
#include <hip/hip_bf16.h>
#include <cstdint>

// ---------------------------------------------------------------------------
// CDNA5 (gfx1250) Mask R-CNN head: ROI-align -> FC (WMMA GEMM) -> cls/box,
// mask branch: ROI-align14 -> 4x conv3x3 (implicit WMMA GEMM) -> deconv2x2
// (WMMA GEMM) -> 1x1 mask conv.
// Matrix math: v_wmma_f32_16x16x32_bf16 (f32 accum).
// Data movement: Tensor Data Mover (tensor_load_to_lds + s_wait_tensorcnt)
// for the wide GEMM tiles; cooperative VMEM->DS staging elsewhere.
// ---------------------------------------------------------------------------

typedef __attribute__((ext_vector_type(16))) __bf16       v16bf;
typedef __attribute__((ext_vector_type(8)))  float        v8f;
typedef __attribute__((ext_vector_type(4)))  unsigned int u32x4;
typedef __attribute__((ext_vector_type(8)))  unsigned int u32x8;

// ------------------------- Tensor Data Mover helper -------------------------
// Issue a 2D tile DMA (global -> LDS) via a D# descriptor:
//   group0: count=1 | lds_addr | global_addr[56:0] | type=2
//   group1: data_size=2B | tensor_dim0/1 | tile_dim0/1 | tensor_dim0_stride
// dims/strides in elements (data_size units); gaddr = byte address of tile
// start; lds_off = LDS byte offset (low 32 bits of the generic pointer).
__device__ __forceinline__ void tdm_load_2d_bf16(uint32_t lds_off,
                                                 const void* gaddr,
                                                 uint32_t tensor_d0,
                                                 uint32_t tensor_d1,
                                                 uint32_t tile_d0,
                                                 uint32_t tile_d1,
                                                 uint64_t stride0) {
  uint64_t ga = (uint64_t)(uintptr_t)gaddr;
  u32x4 g0;
  g0[0] = 1u;                                             // count=1, user mode
  g0[1] = lds_off;                                        // lds_addr
  g0[2] = (uint32_t)ga;                                   // global_addr lo
  g0[3] = (uint32_t)((ga >> 32) & 0x1FFFFFFull) | (2u << 30);  // hi | type=2

  uint64_t q0 = (1ull << 16)                              // data_size = 2B
              | ((uint64_t)(tensor_d0 & 0xFFFFu) << 48);  // dim0[15:0]
  uint64_t q1 = (uint64_t)(tensor_d0 >> 16)               // dim0[31:16]
              | ((uint64_t)tensor_d1 << 16)               // dim1
              | ((uint64_t)(tile_d0 & 0xFFFFu) << 48);    // tile_dim0
  uint64_t q2 = (uint64_t)(tile_d1 & 0xFFFFu)             // tile_dim1
              | ((stride0 & 0xFFFFFFFFull) << 32);        // stride0[31:0]
  uint64_t q3 = (stride0 >> 32) & 0xFFFFull;              // stride0[47:32]
  u32x8 g1;
  g1[0] = (uint32_t)q0; g1[1] = (uint32_t)(q0 >> 32);
  g1[2] = (uint32_t)q1; g1[3] = (uint32_t)(q1 >> 32);
  g1[4] = (uint32_t)q2; g1[5] = (uint32_t)(q2 >> 32);
  g1[6] = (uint32_t)q3; g1[7] = (uint32_t)(q3 >> 32);

  asm volatile("tensor_load_to_lds %0, %1" :: "s"(g0), "s"(g1) : "memory");
}

// --------------------------------- helpers ---------------------------------

__global__ void f32_to_bf16_kernel(const float* __restrict__ in,
                                   __bf16* __restrict__ out, size_t n) {
  size_t i = (size_t)blockIdx.x * blockDim.x + threadIdx.x;
  if (i < n) out[i] = (__bf16)in[i];
}

// b_dec (256) -> expanded bias (1024): col = co*4 + dy*2 + dx
__global__ void expand_bias4_kernel(const float* __restrict__ b,
                                    float* __restrict__ out) {
  int i = blockIdx.x * blockDim.x + threadIdx.x;
  if (i < 1024) out[i] = b[i >> 2];
}

// NCHW bf16 -> [n*HW + p][C] bf16 (GEMM-A layout for the deconv)
__global__ void nchw_to_mk_kernel(const __bf16* __restrict__ in,
                                  __bf16* __restrict__ out,
                                  int NIMG, int C, int HW) {
  size_t i = (size_t)blockIdx.x * blockDim.x + threadIdx.x;
  size_t total = (size_t)NIMG * C * HW;
  if (i >= total) return;
  int p = (int)(i % HW);
  size_t t = i / HW;
  int c = (int)(t % C);
  int n = (int)(t / C);
  out[((size_t)n * HW + p) * C + c] = in[i];
}

// -------------------------------- ROI align --------------------------------
// out[((b*nuse + n)*C + c)*os*os + ph*os + pw]  (bf16)
__global__ void roi_align_kernel(const float* __restrict__ feat,
                                 const float* __restrict__ boxes,
                                 __bf16* __restrict__ out,
                                 int NB, int nboxes_total, int nuse,
                                 int C, int H, int W, int os, int sr,
                                 float scale) {
  size_t idx = (size_t)blockIdx.x * blockDim.x + threadIdx.x;
  size_t total = (size_t)NB * nuse * C * os * os;
  if (idx >= total) return;
  int pw = (int)(idx % os); size_t t = idx / os;
  int ph = (int)(t % os);   t /= os;
  int c  = (int)(t % C);    t /= C;
  int n  = (int)(t % nuse);
  int b  = (int)(t / nuse);

  const float* box = boxes + ((size_t)b * nboxes_total + n) * 4;
  float x1 = box[0] * scale, y1 = box[1] * scale;
  float x2 = box[2] * scale, y2 = box[3] * scale;
  float rw = fmaxf(x2 - x1, 1.0f);
  float rh = fmaxf(y2 - y1, 1.0f);
  const float* f = feat + ((size_t)b * C + c) * H * W;

  float acc = 0.0f;
  for (int sy = 0; sy < sr; ++sy) {
    for (int sx = 0; sx < sr; ++sx) {
      float offy = ((float)(ph * sr + sy) + 0.5f) / (float)sr;
      float offx = ((float)(pw * sr + sx) + 0.5f) / (float)sr;
      float yy = y1 + offy * (rh / (float)os);
      float xx = x1 + offx * (rw / (float)os);
      float y0f = floorf(yy), x0f = floorf(xx);
      float ly = yy - y0f, lx = xx - x0f;
      int y0 = min(max((int)y0f, 0), H - 1);
      int y1i = min(y0 + 1, H - 1);
      int x0 = min(max((int)x0f, 0), W - 1);
      int x1i = min(x0 + 1, W - 1);
      acc += (1.0f - ly) * (1.0f - lx) * f[y0 * W + x0]
           + (1.0f - ly) * lx          * f[y0 * W + x1i]
           + ly          * (1.0f - lx) * f[y1i * W + x0]
           + ly          * lx          * f[y1i * W + x1i];
    }
  }
  out[idx] = (__bf16)(acc / (float)(sr * sr));
}

// ------------------------------ WMMA bf16 GEMM ------------------------------
// C[M,N] = relu?( A[M,K] * B[K,N] + bias[N] )
// Block tile 64x128, 8 waves (wave tile 16x64 = 4 accumulators).
// A row-major bf16, B row-major bf16. M % 64 == 0, K % 32 == 0; N guarded.
// Wide-N case (N % 128 == 0): tiles staged by the Tensor Data Mover.
__global__ __launch_bounds__(256)
void gemm_wmma_bf16_kernel(const __bf16* __restrict__ A,
                           const __bf16* __restrict__ B,
                           const float* __restrict__ bias,
                           float* __restrict__ outF,
                           __bf16* __restrict__ outB,
                           int M, int N, int K, int relu) {
  __shared__ __align__(128) __bf16 smem[64 * 32 + 32 * 128];
  __bf16* sA = smem;             // 64 x 32
  __bf16* sB = smem + 64 * 32;   // 32 x 128

  const int tid  = threadIdx.x;
  const int wid  = tid >> 5;
  const int lane = tid & 31;
  const int wrow = wid >> 1;  // 0..3 (16 rows each)
  const int wcol = wid & 1;   // 0..1 (64 cols each)
  const int m0 = blockIdx.y * 64;
  const int n0 = blockIdx.x * 128;

  v8f acc[4] = {};

  const int ar = tid >> 2, ac = (tid & 3) * 8;    // A: 8 bf16 per thread
  const int br = tid >> 3, bc = (tid & 7) * 16;   // B: 16 bf16 per thread
  const bool use_tdm = ((N & 127) == 0);
  const uint32_t ldsA = (uint32_t)(uintptr_t)sA;  // LDS byte offset
  const uint32_t ldsB = (uint32_t)(uintptr_t)sB;

  for (int k0 = 0; k0 < K; k0 += 32) {
    if (use_tdm) {
      // ---- stage both tiles with the Tensor Data Mover (wave 0 only) ----
      if (wid == 0) {
        // A tile: 32(k) x 64(rows) from A[m0.., k0..]
        tdm_load_2d_bf16(ldsA, A + (size_t)m0 * K + k0,
                         (uint32_t)K, (uint32_t)M, 32u, 64u, (uint64_t)K);
        // B tile: 128(n) x 32(k rows) from B[k0.., n0..]
        tdm_load_2d_bf16(ldsB, B + (size_t)k0 * N + n0,
                         (uint32_t)N, (uint32_t)K, 128u, 32u, (uint64_t)N);
        __builtin_amdgcn_s_wait_tensorcnt(0);
      }
    } else {
      // ---- cooperative staging (narrow N: cls/box heads) ----
      const __bf16* ag = A + (size_t)(m0 + ar) * K + (k0 + ac);
      *(float4*)&sA[ar * 32 + ac] = *(const float4*)ag;
      __builtin_prefetch(ag + 32, 0, 3);   // next A tile -> global_prefetch

      const int gn = n0 + bc;
      const __bf16* bg = B + (size_t)(k0 + br) * N + gn;
      for (int e = 0; e < 16; ++e)
        sB[br * 128 + bc + e] = (gn + e < N) ? bg[e] : (__bf16)0.0f;
    }
    __syncthreads();

    const int hf  = lane >> 4;
    const int l15 = lane & 15;
    v16bf afrag = *(const v16bf*)&sA[(wrow * 16 + l15) * 32 + hf * 16];
    const int kk = l15 + 16 * hf;
#pragma unroll
    for (int j = 0; j < 4; ++j) {
      v16bf bfrag;
      for (int s = 0; s < 16; ++s) bfrag[s] = sB[(16 * hf + s) * 128 + wcol * 64 + j * 16 + l15];
      acc[j] = __builtin_amdgcn_wmma_f32_16x16x32_bf16(
          false, afrag, false, bfrag, (short)0, acc[j], false, false);
    }
    __syncthreads();
  }

  // ---- store (C layout: vgpr r -> M = r + 8*half, lane&15 -> N) ----
  const int hf  = lane >> 4;
  const int l15 = lane & 15;
  for (int j = 0; j < 4; ++j) {
    int gn = n0 + wcol * 64 + j * 16 + l15;
    if (gn >= N) continue;
    float bb = bias ? bias[gn] : 0.0f;
#pragma unroll
    for (int r = 0; r < 8; ++r) {
      int gm = m0 + wrow * 16 + hf * 8 + r;
      float v = acc[j][r] + bb;
      if (relu) v = fmaxf(v, 0.0f);
      if (outF) outF[(size_t)gm * N + gn] = v;
      if (outB) outB[(size_t)gm * N + gn] = (__bf16)v;
    }
  }
}

// --------------------------- conv3x3 implicit GEMM --------------------------
// in/out: [NIMG][256][14][14] bf16 ; w: [256][2304] bf16 (co-major)
// Block: one image, 128 couts x 32 pixels; 8 waves, wave = 16 couts x 32 px.
__global__ __launch_bounds__(256)
void conv3x3_wmma_kernel(const __bf16* __restrict__ in,
                         const __bf16* __restrict__ w,
                         const float* __restrict__ bias,
                         __bf16* __restrict__ out) {
  __shared__ __align__(64) __bf16 sA[128 * 32];  // weight tile
  __shared__ __align__(64) __bf16 sB[32 * 32];   // im2col tile

  const int n   = blockIdx.x;
  const int co0 = blockIdx.y * 128;
  const int p0  = blockIdx.z * 32;
  const int tid = threadIdx.x;
  const int wid = tid >> 5;
  const int lane = tid & 31;

  v8f acc[2] = {};
  const int ar = tid >> 1, ac = (tid & 1) * 16;  // 16 bf16 per thread (A)
  const __bf16* inN = in + (size_t)n * 256 * 196;

  for (int k0 = 0; k0 < 2304; k0 += 32) {
    // weights tile 128x32
    const __bf16* ag = w + (size_t)(co0 + ar) * 2304 + (k0 + ac);
    *(float4*)&sA[ar * 32 + ac]     = *(const float4*)ag;
    *(float4*)&sA[ar * 32 + ac + 8] = *(const float4*)(ag + 8);
    __builtin_prefetch(ag + 32, 0, 3);

    // im2col tile 32x32 (k x pixel), 4 elements per thread
#pragma unroll
    for (int e = 0; e < 4; ++e) {
      int idx = tid * 4 + e;
      int row = idx >> 5, col = idx & 31;
      int k = k0 + row;
      int ci = k / 9, rem = k % 9, dy = rem / 3, dx = rem % 3;
      int p = p0 + col;
      float v = 0.0f;
      if (p < 196) {
        int y = p / 14, x = p % 14;
        int iy = y + dy - 1, ix = x + dx - 1;
        if (iy >= 0 && iy < 14 && ix >= 0 && ix < 14)
          v = (float)inN[ci * 196 + iy * 14 + ix];
      }
      sB[row * 32 + col] = (__bf16)v;
    }
    __syncthreads();

    const int hf  = lane >> 4;
    const int l15 = lane & 15;
    v16bf afrag = *(const v16bf*)&sA[(wid * 16 + l15) * 32 + hf * 16];
    const int kk = l15 + 16 * hf;
#pragma unroll
    for (int j = 0; j < 2; ++j) {
      v16bf bfrag;
      for (int s = 0; s < 16; ++s) bfrag[s] = sB[(16 * hf + s) * 32 + j * 16 + l15];
      acc[j] = __builtin_amdgcn_wmma_f32_16x16x32_bf16(
          false, afrag, false, bfrag, (short)0, acc[j], false, false);
    }
    __syncthreads();
  }

  const int hf  = lane >> 4;
  const int l15 = lane & 15;
  for (int j = 0; j < 2; ++j) {
    int p = p0 + j * 16 + l15;
    if (p >= 196) continue;
#pragma unroll
    for (int r = 0; r < 8; ++r) {
      int co = co0 + wid * 16 + hf * 8 + r;
      float v = fmaxf(acc[j][r] + bias[co], 0.0f);
      out[(size_t)n * 256 * 196 + (size_t)co * 196 + p] = (__bf16)v;
    }
  }
}

// ------------------------------- mask 1x1 conv ------------------------------
// dec layout: [(n*196 + y*14 + x)*1024 + co*4 + dy*2 + dx] bf16 (post ReLU)
// out: [n][k][28][28] f32 at d_out offset
__global__ void mask_head_kernel(const __bf16* __restrict__ dec,
                                 const float* __restrict__ wmsk,
                                 const float* __restrict__ bmsk,
                                 float* __restrict__ out) {
  size_t idx = (size_t)blockIdx.x * blockDim.x + threadIdx.x;
  size_t total = (size_t)128 * 5 * 28 * 28;
  if (idx >= total) return;
  int X = (int)(idx % 28); size_t t = idx / 28;
  int Y = (int)(t % 28);   t /= 28;
  int k = (int)(t % 5);
  int n = (int)(t / 5);
  int y = Y >> 1, dy = Y & 1, x = X >> 1, dx = X & 1;
  const __bf16* row = dec + ((size_t)n * 196 + y * 14 + x) * 1024 + dy * 2 + dx;
  const float* wk = wmsk + (size_t)k * 256;
  float s = bmsk[k];
  for (int co = 0; co < 256; ++co) s += (float)row[co * 4] * wk[co];
  out[idx] = s;
}

// --------------------------------- launcher ---------------------------------

extern "C" void kernel_launch(void* const* d_in, const int* in_sizes, int n_in,
                              void* d_out, int out_size, void* d_ws, size_t ws_size,
                              hipStream_t stream) {
  (void)in_sizes; (void)n_in; (void)out_size; (void)ws_size;

  const float* features = (const float*)d_in[0];   // (2,256,200,304)
  const float* boxes    = (const float*)d_in[1];   // (2,256,4)
  const float* w_fc1    = (const float*)d_in[2];   // (12544,1024)
  const float* b_fc1    = (const float*)d_in[3];
  const float* w_fc2    = (const float*)d_in[4];   // (1024,1024)
  const float* b_fc2    = (const float*)d_in[5];
  const float* w_cls    = (const float*)d_in[6];   // (1024,5)
  const float* b_cls    = (const float*)d_in[7];
  const float* w_box    = (const float*)d_in[8];   // (1024,20)
  const float* b_box    = (const float*)d_in[9];
  const float* w_m[4]   = { (const float*)d_in[10], (const float*)d_in[12],
                            (const float*)d_in[14], (const float*)d_in[16] };
  const float* b_m[4]   = { (const float*)d_in[11], (const float*)d_in[13],
                            (const float*)d_in[15], (const float*)d_in[17] };
  const float* w_dec    = (const float*)d_in[18];  // (256,256,2,2)
  const float* b_dec    = (const float*)d_in[19];
  const float* w_msk    = (const float*)d_in[20];  // (5,256,1,1)
  const float* b_msk    = (const float*)d_in[21];

  float* out = (float*)d_out;      // cls 2560 | box 10240 | mask 501760

  // ---- workspace carve-up ----
  char* base = (char*)d_ws;
  size_t off = 0;
  auto take = [&](size_t bytes) -> char* {
    char* p = base + off;
    off += (bytes + 255) & ~(size_t)255;
    return p;
  };
  __bf16* A_fc    = (__bf16*)take((size_t)512 * 12544 * 2);
  __bf16* wfc1_bf = (__bf16*)take((size_t)12544 * 1024 * 2);
  __bf16* fc1_bf  = (__bf16*)take((size_t)512 * 1024 * 2);
  __bf16* wfc2_bf = (__bf16*)take((size_t)1024 * 1024 * 2);
  __bf16* fc2_bf  = (__bf16*)take((size_t)512 * 1024 * 2);
  __bf16* wcls_bf = (__bf16*)take((size_t)1024 * 5 * 2);
  __bf16* wbox_bf = (__bf16*)take((size_t)1024 * 20 * 2);
  __bf16* mroi_bf = (__bf16*)take((size_t)128 * 256 * 196 * 2);
  __bf16* wm_bf[4];
  for (int i = 0; i < 4; ++i) wm_bf[i] = (__bf16*)take((size_t)256 * 2304 * 2);
  __bf16* buf0    = (__bf16*)take((size_t)128 * 256 * 196 * 2);
  __bf16* buf1    = (__bf16*)take((size_t)128 * 256 * 196 * 2);
  __bf16* nhwc_bf = (__bf16*)take((size_t)128 * 196 * 256 * 2);
  __bf16* wdec_bf = (__bf16*)take((size_t)256 * 1024 * 2);
  float*  bdec_x4 = (float*)take((size_t)1024 * 4);
  __bf16* dec_bf  = (__bf16*)take((size_t)25088 * 1024 * 2);

  auto cvt = [&](const float* src, __bf16* dst, size_t n) {
    int blocks = (int)((n + 255) / 256);
    hipLaunchKernelGGL(f32_to_bf16_kernel, dim3(blocks), dim3(256), 0, stream,
                       src, dst, n);
  };

  // ---- box head ----
  {
    size_t total = (size_t)2 * 256 * 256 * 7 * 7;
    hipLaunchKernelGGL(roi_align_kernel, dim3((total + 255) / 256), dim3(256),
                       0, stream, features, boxes, A_fc,
                       2, 256, 256, 256, 200, 304, 7, 2, 0.25f);
  }
  cvt(w_fc1, wfc1_bf, (size_t)12544 * 1024);
  hipLaunchKernelGGL(gemm_wmma_bf16_kernel, dim3(1024 / 128, 512 / 64),
                     dim3(256), 0, stream, A_fc, wfc1_bf, b_fc1,
                     (float*)nullptr, fc1_bf, 512, 1024, 12544, 1);
  cvt(w_fc2, wfc2_bf, (size_t)1024 * 1024);
  hipLaunchKernelGGL(gemm_wmma_bf16_kernel, dim3(1024 / 128, 512 / 64),
                     dim3(256), 0, stream, fc1_bf, wfc2_bf, b_fc2,
                     (float*)nullptr, fc2_bf, 512, 1024, 1024, 1);
  cvt(w_cls, wcls_bf, (size_t)1024 * 5);
  hipLaunchKernelGGL(gemm_wmma_bf16_kernel, dim3(1, 512 / 64),
                     dim3(256), 0, stream, fc2_bf, wcls_bf, b_cls,
                     out + 0, (__bf16*)nullptr, 512, 5, 1024, 0);
  cvt(w_box, wbox_bf, (size_t)1024 * 20);
  hipLaunchKernelGGL(gemm_wmma_bf16_kernel, dim3(1, 512 / 64),
                     dim3(256), 0, stream, fc2_bf, wbox_bf, b_box,
                     out + 2560, (__bf16*)nullptr, 512, 20, 1024, 0);

  // ---- mask branch ----
  {
    size_t total = (size_t)2 * 64 * 256 * 14 * 14;
    hipLaunchKernelGGL(roi_align_kernel, dim3((total + 255) / 256), dim3(256),
                       0, stream, features, boxes, mroi_bf,
                       2, 256, 64, 256, 200, 304, 14, 1, 0.25f);
  }
  for (int i = 0; i < 4; ++i) cvt(w_m[i], wm_bf[i], (size_t)256 * 2304);

  const __bf16* cin = mroi_bf;
  __bf16* cout_bufs[4] = { buf0, buf1, buf0, buf1 };
  for (int i = 0; i < 4; ++i) {
    hipLaunchKernelGGL(conv3x3_wmma_kernel, dim3(128, 2, 7), dim3(256),
                       0, stream, cin, wm_bf[i], b_m[i], cout_bufs[i]);
    cin = cout_bufs[i];
  }

  {
    size_t total = (size_t)128 * 256 * 196;
    hipLaunchKernelGGL(nchw_to_mk_kernel, dim3((total + 255) / 256), dim3(256),
                       0, stream, cin, nhwc_bf, 128, 256, 196);
  }
  cvt(w_dec, wdec_bf, (size_t)256 * 1024);
  hipLaunchKernelGGL(expand_bias4_kernel, dim3(4), dim3(256), 0, stream,
                     b_dec, bdec_x4);
  // deconv as GEMM: M = 128*196 = 25088, K = 256, N = 1024 (co,dy,dx)
  hipLaunchKernelGGL(gemm_wmma_bf16_kernel, dim3(1024 / 128, 25088 / 64),
                     dim3(256), 0, stream, nhwc_bf, wdec_bf, bdec_x4,
                     (float*)nullptr, dec_bf, 25088, 1024, 256, 1);

  {
    size_t total = (size_t)128 * 5 * 28 * 28;
    hipLaunchKernelGGL(mask_head_kernel, dim3((total + 255) / 256), dim3(256),
                       0, stream, dec_bf, w_msk, b_msk, out + 12800);
  }
}
